// StreamingFusionGatingTransformer_33724083208867
// MI455X (gfx1250) — matrix operand order repaired and checked by reading, not compiled, not run on hardware
//
#include <hip/hip_runtime.h>
#include <hip/hip_bf16.h>
#include <math.h>

// ---------------- problem constants ----------------
constexpr int NB  = 32;     // batch
constexpr int NT  = 32;     // tokens
constexpr int ND  = 512;    // model dim
constexpr int NH  = 8;      // heads
constexpr int NHD = 64;     // head dim
constexpr int NV  = 49;     // visual tokens
constexpr int NC  = 2048;   // cache capacity
constexpr int NSQ = 33;     // query rows (1+T)
constexpr int NKEYS = NV + NC + NT;  // 2129
constexpr int NKP  = 2176;           // keys padded to 68*32
constexpr float kScale = 0.125f;     // HD^-0.5

typedef __attribute__((ext_vector_type(16))) __bf16 v16bf;
typedef __attribute__((ext_vector_type(8)))  __bf16 v8bf;
typedef __attribute__((ext_vector_type(4)))  __bf16 v4bf;
typedef __attribute__((ext_vector_type(8)))  float  v8f;

__device__ __forceinline__ v16bf cat16(v8bf a, v8bf b) {
  v16bf r;
#pragma unroll
  for (int e = 0; e < 8; ++e) { r[e] = a[e]; r[8 + e] = b[e]; }
  return r;
}

// ---------------- elementwise helpers (float4 vectorized) ----------------
__global__ void build_combined_kernel(const float* __restrict__ prev_cls,
                                      const float* __restrict__ cur_tok,
                                      float* __restrict__ combined, int n4) {
  int i = blockIdx.x * blockDim.x + threadIdx.x;
  if (i >= n4) return;
  int row = i >> 7, d4 = i & 127;          // 128 float4 per 512-row
  int b = row / NSQ, s = row % NSQ;
  const float4* src = (s == 0)
      ? (const float4*)(prev_cls + (size_t)b * ND) + d4
      : (const float4*)(cur_tok + ((size_t)(b * NT + s - 1)) * ND) + d4;
  ((float4*)combined)[i] = *src;
}

// weight convert + transpose: W[K,N] f32 -> Wt[N,K] bf16 (k-contiguous rows)
__global__ void cvt_wt_kernel(const float* __restrict__ W, __bf16* __restrict__ Wt,
                              int K, int N, int n8) {
  int i = blockIdx.x * blockDim.x + threadIdx.x;
  if (i >= n8) return;
  int kgrp = i % (K / 8);
  int n = i / (K / 8);
  int k0 = kgrp * 8;
  v8bf o;
#pragma unroll
  for (int j = 0; j < 8; ++j) o[j] = (__bf16)W[(size_t)(k0 + j) * N + n];
  *(v8bf*)(Wt + (size_t)n * K + k0) = o;
}

__global__ void add_inplace_kernel(float* __restrict__ y,
                                   const float* __restrict__ x, int n4) {
  int i = blockIdx.x * blockDim.x + threadIdx.x;
  if (i >= n4) return;
  float4 a = ((float4*)y)[i];
  float4 b = ((const float4*)x)[i];
  a.x += b.x; a.y += b.y; a.z += b.z; a.w += b.w;
  ((float4*)y)[i] = a;
}

__global__ void gelu_cvt_kernel(const float* __restrict__ x,
                                __bf16* __restrict__ y, int n4) {
  int i = blockIdx.x * blockDim.x + threadIdx.x;
  if (i >= n4) return;
  float4 v = ((const float4*)x)[i];
  v4bf o;
  o[0] = (__bf16)(0.5f * v.x * (1.0f + erff(v.x * 0.70710678118654752f)));
  o[1] = (__bf16)(0.5f * v.y * (1.0f + erff(v.y * 0.70710678118654752f)));
  o[2] = (__bf16)(0.5f * v.z * (1.0f + erff(v.z * 0.70710678118654752f)));
  o[3] = (__bf16)(0.5f * v.w * (1.0f + erff(v.w * 0.70710678118654752f)));
  *(v4bf*)(y + (size_t)i * 4) = o;
}

// ---------------- LayerNorm (one 512-row per block, 128 threads) ----------------
__global__ __launch_bounds__(128)
void ln_kernel(const float* __restrict__ x, const float* __restrict__ g,
               const float* __restrict__ bta, float* __restrict__ outF,
               __bf16* __restrict__ outH) {
  __shared__ float red[8];
  const int row = blockIdx.x, tid = threadIdx.x;
  float4 v = ((const float4*)(x + (size_t)row * ND))[tid];
  float s = v.x + v.y + v.z + v.w;
  float ss = v.x * v.x + v.y * v.y + v.z * v.z + v.w * v.w;
#pragma unroll
  for (int m = 16; m >= 1; m >>= 1) {
    s += __shfl_xor(s, m, 32);
    ss += __shfl_xor(ss, m, 32);
  }
  const int wave = tid >> 5, lane = tid & 31;
  if (lane == 0) { red[wave] = s; red[4 + wave] = ss; }
  __syncthreads();
  s = red[0] + red[1] + red[2] + red[3];
  ss = red[4] + red[5] + red[6] + red[7];
  const float mu = s * (1.0f / ND);
  const float rr = rsqrtf(ss * (1.0f / ND) - mu * mu + 1e-5f);
  float4 gg = ((const float4*)g)[tid];
  float4 bb = ((const float4*)bta)[tid];
  float4 y;
  y.x = (v.x - mu) * rr * gg.x + bb.x;
  y.y = (v.y - mu) * rr * gg.y + bb.y;
  y.z = (v.z - mu) * rr * gg.z + bb.z;
  y.w = (v.w - mu) * rr * gg.w + bb.w;
  if (outF) ((float4*)(outF + (size_t)row * ND))[tid] = y;
  if (outH) {
    v4bf o;
    o[0] = (__bf16)y.x; o[1] = (__bf16)y.y; o[2] = (__bf16)y.z; o[3] = (__bf16)y.w;
    *(v4bf*)(outH + (size_t)row * ND + tid * 4) = o;
  }
}

// ---------------- WMMA bf16 GEMM: C[M,N] = A[M,K] @ Wt[N,K]^T + bias ----------------
// tile 32(M) x 64(N), 128 threads = 4 waves, each wave one 16x32 quadrant.
// A tile: double-buffered LDS via GLOBAL_LOAD_ASYNC_TO_LDS_B128 (ASYNCcnt DMA).
// B fragments: streamed straight from L2-resident transposed weights (b128 loads),
// manually 2x-unrolled software pipeline (two named register sets, no rotation moves).
// Hand-rolled barrier waits only ASYNCcnt+DScnt, NOT LOADcnt, so the in-flight
// next-step B loads are not flushed at the barrier.
__global__ __launch_bounds__(128)
void wmma_gemm_kernel(const __bf16* __restrict__ A, const __bf16* __restrict__ Wt,
                      const float* __restrict__ bias, float* __restrict__ outF,
                      __bf16* __restrict__ outH, int M, int N, int K) {
  __shared__ __bf16 As[2][32 * 40];   // [m][k], rows padded to 80 B
  const int tid = threadIdx.x;
  const int lane = tid & 31;
  const int wave = tid >> 5;
  const int lo = lane & 15, hi = lane >> 4;
  const int rowBlk = blockIdx.x * 32;
  const int colBlk = blockIdx.y * 64;
  const int wrt = wave >> 1;  // row tile 0..1
  const int wct = wave & 1;   // col pair 0..1
  const uint64_t abase = (uint64_t)(uintptr_t)A;
  const int arow = tid >> 2, aseg = tid & 3;

  auto stageA = [&](int buf, int k0) {
    // per-lane async DMA: 16 B global -> LDS, tracked by ASYNCcnt
    uint32_t lds = (uint32_t)(uintptr_t)(&As[buf][arow * 40 + aseg * 8]);
    uint32_t voff = (uint32_t)(((rowBlk + arow) * K + k0 + aseg * 8) * 2);
    asm volatile("global_load_async_to_lds_b128 %0, %1, %2"
                 :: "v"(lds), "v"(voff), "s"(abase) : "memory");
  };
  auto barrier_light = [&]() {
    // async DMA done + our LDS reads retired; do NOT flush LOADcnt (B prefetch)
    asm volatile("s_wait_asynccnt 0x0\n\t"
                 "s_wait_dscnt 0x0\n\t"
                 "s_barrier_signal -1\n\t"
                 "s_barrier_wait -1" ::: "memory");
  };

  const __bf16* w0r = Wt + (size_t)(colBlk + wct * 32 + lo) * K;        // col c0
  const __bf16* w1r = Wt + (size_t)(colBlk + wct * 32 + 16 + lo) * K;   // col c1
  auto loadB = [&](int k0, v16bf& b0, v16bf& b1) {
    const int kn = k0 + hi * 16;
    b0 = cat16(*(const v8bf*)(w0r + kn), *(const v8bf*)(w0r + kn + 8));
    b1 = cat16(*(const v8bf*)(w1r + kn), *(const v8bf*)(w1r + kn + 8));
  };

  v8f acc0 = {}, acc1 = {};
  auto compute = [&](int buf, v16bf b0, v16bf b1) {
    const __bf16* ar = &As[buf][(wrt * 16 + lo) * 40];
    v16bf a = cat16(*(const v8bf*)(ar + hi * 16), *(const v8bf*)(ar + hi * 16 + 8));
    acc0 = __builtin_amdgcn_wmma_f32_16x16x32_bf16(false, a, false, b0, (short)0, acc0, false, false);
    acc1 = __builtin_amdgcn_wmma_f32_16x16x32_bf16(false, a, false, b1, (short)0, acc1, false, false);
  };

  v16bf b0A, b1A, b0B, b1B;
  stageA(0, 0);
  loadB(0, b0A, b1A);
  barrier_light();

  const int nsteps = K / 32;  // even for all K used here (16 or 64)
  for (int step = 0; step < nsteps; step += 2) {
    stageA(1, (step + 1) * 32);         // DMA overlaps WMMAs below
    loadB((step + 1) * 32, b0B, b1B);   // register prefetch, 1 step ahead
    compute(0, b0A, b1A);
    barrier_light();
    if (step + 2 < nsteps) {
      stageA(0, (step + 2) * 32);
      loadB((step + 2) * 32, b0A, b1A);
    }
    compute(1, b0B, b1B);
    barrier_light();
  }

#pragma unroll
  for (int r = 0; r < 8; ++r) {
    int row = rowBlk + wrt * 16 + r + hi * 8;
    int c0 = colBlk + wct * 32 + lo;
    int c1 = c0 + 16;
    float v0 = acc0[r] + (bias ? bias[c0] : 0.f);
    float v1 = acc1[r] + (bias ? bias[c1] : 0.f);
    if (row < M) {
      if (outF) { outF[(size_t)row * N + c0] = v0; outF[(size_t)row * N + c1] = v1; }
      if (outH) { outH[(size_t)row * N + c0] = (__bf16)v0; outH[(size_t)row * N + c1] = (__bf16)v1; }
    }
  }
}

// ---------------- dense K/V staging for attention ----------------
__device__ __forceinline__ float fetch_src(int b, int h, int s, int d,
                                           const __bf16* __restrict__ vis,
                                           const __bf16* __restrict__ cur,
                                           const float* __restrict__ cache) {
  if (s < NV)
    return (float)vis[((size_t)(b * NV + s)) * ND + h * NHD + d];
  if (s < NV + NC)
    return cache[(((size_t)(b * NH + h)) * NC + (s - NV)) * NHD + d];
  if (s < NKEYS)
    return (float)cur[((size_t)(b * NSQ + (s - NV - NC) + 1)) * ND + h * NHD + d];
  return 0.f;
}

// Kb[b][h][s(NKP)][d(64)] bf16 — row-major keys, contiguous head dim
__global__ void kprep_kernel(const __bf16* __restrict__ vis, const __bf16* __restrict__ cur,
                             const float* __restrict__ cache, __bf16* __restrict__ Kb, int n8) {
  int i = blockIdx.x * blockDim.x + threadIdx.x;
  if (i >= n8) return;
  int dgrp = i & 7;                 // 8 groups of 8 dims
  int tmp = i >> 3;
  int s = tmp % NKP;
  int bh = tmp / NKP;
  int b = bh >> 3, h = bh & 7;
  int d0 = dgrp * 8;
  v8bf o;
  if (s < NV) {
    o = *(const v8bf*)(vis + ((size_t)(b * NV + s)) * ND + h * NHD + d0);
  } else if (s < NV + NC) {
    const float* p = cache + (((size_t)(b * NH + h)) * NC + (s - NV)) * NHD + d0;
#pragma unroll
    for (int j = 0; j < 8; ++j) o[j] = (__bf16)p[j];
  } else if (s < NKEYS) {
    o = *(const v8bf*)(cur + ((size_t)(b * NSQ + (s - NV - NC) + 1)) * ND + h * NHD + d0);
  } else {
    o = (v8bf)(__bf16)0.f;
  }
  *(v8bf*)(Kb + (size_t)i * 8) = o;
}

// Vt[b][h][d(64)][s(NKP)] bf16 — transposed values for contiguous-key loads
__global__ void vtprep_kernel(const __bf16* __restrict__ vis, const __bf16* __restrict__ cur,
                              const float* __restrict__ cache, __bf16* __restrict__ Vt, int n8) {
  int i = blockIdx.x * blockDim.x + threadIdx.x;
  if (i >= n8) return;
  int sgrp = i % (NKP / 8);
  int rest = i / (NKP / 8);
  int d = rest & 63;
  int bh = rest >> 6;
  int b = bh >> 3, h = bh & 7;
  int s0 = sgrp * 8;
  v8bf o;
#pragma unroll
  for (int j = 0; j < 8; ++j) o[j] = (__bf16)fetch_src(b, h, s0 + j, d, vis, cur, cache);
  *(v8bf*)(Vt + (size_t)i * 8) = o;
}

__device__ __forceinline__ bool key_valid(int s, int myLen,
                                          const int* __restrict__ tmask, int b) {
  if (s < NV) return true;
  if (s < NV + NC) return (s - NV) < myLen;
  if (s < NKEYS) return tmask[b * NT + (s - NV - NC)] != 0;
  return false;
}

// ---------------- flash attention: 1 block per (b,h), 3 waves = 3 q-tiles ----------------
__global__ __launch_bounds__(96)
void attn_flash_kernel(const __bf16* __restrict__ qh, const __bf16* __restrict__ Kb,
                       const __bf16* __restrict__ Vt, const int* __restrict__ clen,
                       const int* __restrict__ tmask, __bf16* __restrict__ attnH) {
  __shared__ __bf16 Ps[3][16 * 32];
  const int b = blockIdx.x >> 3, h = blockIdx.x & 7;
  const int lane = threadIdx.x & 31, wave = threadIdx.x >> 5;
  const int lo = lane & 15, hi = lane >> 4;
  const int qrow0 = wave * 16;
  const size_t bhK = ((size_t)(b * NH + h)) * NKP;   // key-row base (x64 dims)
  const size_t bhV = ((size_t)(b * NH + h)) * NHD;   // value-row base (xNKP keys)

  // q fragments (A layout): one contiguous 16-elem run per chunk -> b128 loads
  v16bf qa0 = {}, qa1 = {};
  {
    const int srow = qrow0 + lo;
    if (srow < NSQ) {
      const __bf16* qr = qh + ((size_t)(b * NSQ + srow)) * ND + h * NHD;
      qa0 = cat16(*(const v8bf*)(qr + hi * 16), *(const v8bf*)(qr + hi * 16 + 8));
      qa1 = cat16(*(const v8bf*)(qr + 32 + hi * 16), *(const v8bf*)(qr + 40 + hi * 16));
    }
  }

  float m8[8], l8[8];
  v8f O0 = {}, O1 = {}, O2 = {}, O3 = {};
#pragma unroll
  for (int r = 0; r < 8; ++r) { m8[r] = -1e30f; l8[r] = 0.f; }
  const int myLen = clen[b];

  for (int kb = 0; kb < NKEYS; kb += 32) {
    v8f s0 = {}, s1 = {};
    bool kval0, kval1;
    {  // col tile 0: keys kb+lo
      int s = kb + lo;
      kval0 = key_valid(s, myLen, tmask, b);
      const __bf16* kp = Kb + (bhK + s) * NHD;
      v16bf bk0 = cat16(*(const v8bf*)(kp + hi * 16), *(const v8bf*)(kp + hi * 16 + 8));
      v16bf bk1 = cat16(*(const v8bf*)(kp + 32 + hi * 16), *(const v8bf*)(kp + 32 + hi * 16 + 8));
      s0 = __builtin_amdgcn_wmma_f32_16x16x32_bf16(false, qa0, false, bk0, (short)0, s0, false, false);
      s0 = __builtin_amdgcn_wmma_f32_16x16x32_bf16(false, qa1, false, bk1, (short)0, s0, false, false);
    }
    {  // col tile 1: keys kb+16+lo
      int s = kb + 16 + lo;
      kval1 = key_valid(s, myLen, tmask, b);
      const __bf16* kp = Kb + (bhK + s) * NHD;
      v16bf bk0 = cat16(*(const v8bf*)(kp + hi * 16), *(const v8bf*)(kp + hi * 16 + 8));
      v16bf bk1 = cat16(*(const v8bf*)(kp + 32 + hi * 16), *(const v8bf*)(kp + 32 + hi * 16 + 8));
      s1 = __builtin_amdgcn_wmma_f32_16x16x32_bf16(false, qa0, false, bk0, (short)0, s1, false, false);
      s1 = __builtin_amdgcn_wmma_f32_16x16x32_bf16(false, qa1, false, bk1, (short)0, s1, false, false);
    }

    // online softmax update per row (row = r + 8*hi in this wave's q tile)
#pragma unroll
    for (int r = 0; r < 8; ++r) {
      float sc0 = kval0 ? s0[r] * kScale : -1e30f;
      float sc1 = kval1 ? s1[r] * kScale : -1e30f;
      float bm = fmaxf(sc0, sc1);
#pragma unroll
      for (int mk = 8; mk >= 1; mk >>= 1) bm = fmaxf(bm, __shfl_xor(bm, mk, 32));
      float nm = fmaxf(m8[r], bm);
      float esc = __expf(m8[r] - nm);
      float p0 = __expf(sc0 - nm);
      float p1 = __expf(sc1 - nm);
      float rs = p0 + p1;
#pragma unroll
      for (int mk = 8; mk >= 1; mk >>= 1) rs += __shfl_xor(rs, mk, 32);
      l8[r] = l8[r] * esc + rs;
      m8[r] = nm;
      O0[r] *= esc; O1[r] *= esc; O2[r] *= esc; O3[r] *= esc;
      int prow = r + hi * 8;
      Ps[wave][prow * 32 + lo] = (__bf16)p0;
      Ps[wave][prow * 32 + 16 + lo] = (__bf16)p1;
    }
    // per-wave LDS round trip: C-layout P -> A-layout fragment (b128 loads)
    asm volatile("s_wait_dscnt 0x0" ::: "memory");
    const __bf16* pr = &Ps[wave][lo * 32 + hi * 16];
    v16bf pa = cat16(*(const v8bf*)pr, *(const v8bf*)(pr + 8));

#pragma unroll
    for (int ot = 0; ot < 4; ++ot) {
      int d = ot * 16 + lo;
      const __bf16* vp = Vt + (bhV + d) * NKP + kb + hi * 16;
      v16bf vb = cat16(*(const v8bf*)vp, *(const v8bf*)(vp + 8));
      v8f* Op = (ot == 0) ? &O0 : (ot == 1) ? &O1 : (ot == 2) ? &O2 : &O3;
      *Op = __builtin_amdgcn_wmma_f32_16x16x32_bf16(false, pa, false, vb, (short)0, *Op, false, false);
    }
  }

#pragma unroll
  for (int r = 0; r < 8; ++r) {
    int row = qrow0 + r + hi * 8;
    if (row < NSQ) {
      float inv = 1.0f / l8[r];
      size_t base = ((size_t)(b * NSQ + row)) * ND + h * NHD;
      attnH[base + 0 + lo]  = (__bf16)(O0[r] * inv);
      attnH[base + 16 + lo] = (__bf16)(O1[r] * inv);
      attnH[base + 32 + lo] = (__bf16)(O2[r] * inv);
      attnH[base + 48 + lo] = (__bf16)(O3[r] * inv);
    }
  }
}

// ---------------- finalize: FFN residual, mask, cls select, write out ----------------
__global__ void finalize_kernel(const float* __restrict__ combined,
                                const float* __restrict__ ffn2,
                                const float* __restrict__ prev_cls,
                                const int* __restrict__ tmask,
                                const int* __restrict__ act,
                                float* __restrict__ out,
                                float* __restrict__ ntok, int n4) {
  int i = blockIdx.x * blockDim.x + threadIdx.x;
  if (i >= n4) return;
  int row = i >> 7, d4 = i & 127;
  int b = row / NSQ, s = row % NSQ;
  float4 c = ((const float4*)combined)[i];
  float4 f = ((const float4*)ffn2)[i];
  float4 v;
  v.x = c.x + f.x; v.y = c.y + f.y; v.z = c.z + f.z; v.w = c.w + f.w;
  if (s == 0) {
    ((float4*)out)[i] = act[b] ? v : ((const float4*)(prev_cls + (size_t)b * ND))[d4];
  } else {
    if (!tmask[b * NT + s - 1]) v.x = v.y = v.z = v.w = 0.f;
    ((float4*)out)[i] = v;
    ((float4*)(ntok + ((size_t)(b * NT + s - 1)) * ND))[d4] = v;
  }
}

// ---------------- cache append: destination indices + new_len ----------------
__global__ __launch_bounds__(32)
void dest_kernel(const int* __restrict__ tmask, const int* __restrict__ act,
                 const int* __restrict__ clen, int* __restrict__ dest,
                 float* __restrict__ lenOut) {
  int b = blockIdx.x, t = threadIdx.x;
  int a = act[b];
  int c = 0;
  for (int j = 0; j <= t; ++j) c += (tmask[b * NT + j] && a) ? 1 : 0;
  int msk = tmask[b * NT + t] && a;
  dest[b * NT + t] = msk ? (clen[b] + c - 1) : NC;
  if (t == NT - 1) lenOut[b] = (float)(clen[b] + c);
}

__global__ __launch_bounds__(128)
void scatter_kernel(const float* __restrict__ kapp, const float* __restrict__ vapp,
                    const int* __restrict__ dest, float* __restrict__ newK,
                    float* __restrict__ newV) {
  int bt = blockIdx.x;
  int b = bt >> 5;
  int dd = dest[bt];
  if (dd < 0 || dd >= NC) return;  // dropped rows
  int i4 = threadIdx.x;            // 128 float4 per row of 512
  int h = i4 >> 4;
  int off4 = i4 & 15;
  float4 kv = ((const float4*)(kapp + (size_t)bt * ND))[i4];
  float4 vv = ((const float4*)(vapp + (size_t)bt * ND))[i4];
  size_t dst4 = ((((size_t)(b * NH + h)) * NC + dd) * NHD) / 4 + off4;
  ((float4*)newK)[dst4] = kv;
  ((float4*)newV)[dst4] = vv;
}

// ---------------- host launch ----------------
extern "C" void kernel_launch(void* const* d_in, const int* in_sizes, int n_in,
                              void* d_out, int out_size, void* d_ws, size_t ws_size,
                              hipStream_t stream) {
  const float* prev_cls = (const float*)d_in[0];
  const float* cur_tok  = (const float*)d_in[1];
  const float* vis_tok  = (const float*)d_in[2];
  const float* cacheK   = (const float*)d_in[3];
  const float* cacheV   = (const float*)d_in[4];
  const float* Wq = (const float*)d_in[5];  const float* bq = (const float*)d_in[6];
  const float* Wk = (const float*)d_in[7];  const float* bk = (const float*)d_in[8];
  const float* Wv = (const float*)d_in[9];  const float* bv = (const float*)d_in[10];
  const float* Wo = (const float*)d_in[11]; const float* bo = (const float*)d_in[12];
  const float* W1 = (const float*)d_in[13]; const float* b1 = (const float*)d_in[14];
  const float* W2 = (const float*)d_in[15]; const float* b2 = (const float*)d_in[16];
  const float* lnqg = (const float*)d_in[17]; const float* lnqb = (const float*)d_in[18];
  const float* lnvg = (const float*)d_in[19]; const float* lnvb = (const float*)d_in[20];
  const float* lnfg = (const float*)d_in[21]; const float* lnfb = (const float*)d_in[22];
  const float* lncg = (const float*)d_in[23]; const float* lncb = (const float*)d_in[24];
  const int* tmask = (const int*)d_in[25];
  const int* clen  = (const int*)d_in[26];
  const int* act   = (const int*)d_in[27];

  const size_t nOut   = (size_t)NB * NSQ * ND;          // 540672
  const size_t nCache = (size_t)NB * NH * NC * NHD;     // 33554432
  float* out  = (float*)d_out;
  float* newK = out + nOut;
  float* newV = newK + nCache;
  float* lenO = newV + nCache;

  // workspace carve-out
  char* wp = (char*)d_ws;
  size_t off = 0;
  auto alloc = [&](size_t bytes) -> void* {
    void* r = wp + off;
    off += (bytes + 255) & ~(size_t)255;
    return r;
  };
  const int MQ = NB * NSQ;  // 1056
  const int MV = NB * NV;   // 1568
  const int MT = NB * NT;   // 1024
  float*  combined = (float*)alloc((size_t)MQ * ND * 4);
  __bf16* qin_h    = (__bf16*)alloc((size_t)MQ * ND * 2);
  __bf16* vis_h    = (__bf16*)alloc((size_t)MV * ND * 2);
  __bf16* WqT = (__bf16*)alloc((size_t)ND * ND * 2);       // [N][K] transposed bf16
  __bf16* WkT = (__bf16*)alloc((size_t)ND * ND * 2);
  __bf16* WvT = (__bf16*)alloc((size_t)ND * ND * 2);
  __bf16* WoT = (__bf16*)alloc((size_t)ND * ND * 2);
  __bf16* W1T = (__bf16*)alloc((size_t)ND * 4 * ND * 2);   // [4*ND][ND]
  __bf16* W2T = (__bf16*)alloc((size_t)4 * ND * ND * 2);   // [ND][4*ND]
  __bf16* qH    = (__bf16*)alloc((size_t)MQ * ND * 2);
  __bf16* kcurH = (__bf16*)alloc((size_t)MQ * ND * 2);
  __bf16* vcurH = (__bf16*)alloc((size_t)MQ * ND * 2);
  __bf16* kvisH = (__bf16*)alloc((size_t)MV * ND * 2);
  __bf16* vvisH = (__bf16*)alloc((size_t)MV * ND * 2);
  __bf16* Kb    = (__bf16*)alloc((size_t)NB * NH * NKP * NHD * 2);
  __bf16* Vt    = (__bf16*)alloc((size_t)NB * NH * NHD * NKP * 2);
  __bf16* attnHb = (__bf16*)alloc((size_t)MQ * ND * 2);
  float*  attnProj = (float*)alloc((size_t)MQ * ND * 4);
  __bf16* hBf   = (__bf16*)alloc((size_t)MQ * ND * 2);
  float*  ffn1  = (float*)alloc((size_t)MQ * 4 * ND * 4);
  __bf16* ffn1H = (__bf16*)alloc((size_t)MQ * 4 * ND * 2);
  float*  ffn2  = (float*)alloc((size_t)MQ * ND * 4);
  float*  ntok  = (float*)alloc((size_t)MT * ND * 4);
  __bf16* clnH  = (__bf16*)alloc((size_t)MT * ND * 2);
  float*  kapp  = (float*)alloc((size_t)MT * ND * 4);
  float*  vapp  = (float*)alloc((size_t)MT * ND * 4);
  int*    dest  = (int*)alloc((size_t)NB * NT * 4);
  (void)ws_size; (void)n_in; (void)in_sizes; (void)out_size;

  const int EW = 256;
  auto blocks = [&](size_t n) { return (unsigned)((n + EW - 1) / EW); };

  // 1) combined + layernorms
  build_combined_kernel<<<blocks(nOut / 4), EW, 0, stream>>>(prev_cls, cur_tok, combined, (int)(nOut / 4));
  ln_kernel<<<MQ, 128, 0, stream>>>(combined, lnqg, lnqb, nullptr, qin_h);
  ln_kernel<<<MV, 128, 0, stream>>>(vis_tok, lnvg, lnvb, nullptr, vis_h);

  // 2) weights -> transposed bf16 (Wt[n][k], k contiguous)
  cvt_wt_kernel<<<blocks(ND * ND / 8), EW, 0, stream>>>(Wq, WqT, ND, ND, ND * ND / 8);
  cvt_wt_kernel<<<blocks(ND * ND / 8), EW, 0, stream>>>(Wk, WkT, ND, ND, ND * ND / 8);
  cvt_wt_kernel<<<blocks(ND * ND / 8), EW, 0, stream>>>(Wv, WvT, ND, ND, ND * ND / 8);
  cvt_wt_kernel<<<blocks(ND * ND / 8), EW, 0, stream>>>(Wo, WoT, ND, ND, ND * ND / 8);
  cvt_wt_kernel<<<blocks(ND * ND / 2), EW, 0, stream>>>(W1, W1T, ND, 4 * ND, ND * ND / 2);
  cvt_wt_kernel<<<blocks(ND * ND / 2), EW, 0, stream>>>(W2, W2T, 4 * ND, ND, ND * ND / 2);

  // 3) Q/K/V projections (WMMA bf16)
  dim3 g512(MQ / 32, ND / 64);
  wmma_gemm_kernel<<<g512, 128, 0, stream>>>(qin_h, WqT, bq, nullptr, qH, MQ, ND, ND);
  wmma_gemm_kernel<<<g512, 128, 0, stream>>>(qin_h, WkT, bk, nullptr, kcurH, MQ, ND, ND);
  wmma_gemm_kernel<<<g512, 128, 0, stream>>>(qin_h, WvT, bv, nullptr, vcurH, MQ, ND, ND);
  dim3 gvis(MV / 32, ND / 64);
  wmma_gemm_kernel<<<gvis, 128, 0, stream>>>(vis_h, WkT, bk, nullptr, kvisH, MV, ND, ND);
  wmma_gemm_kernel<<<gvis, 128, 0, stream>>>(vis_h, WvT, bv, nullptr, vvisH, MV, ND, ND);

  // 4) dense K / transposed V staging, then flash attention
  {
    int n8k = NB * NH * NKP * NHD / 8;
    kprep_kernel<<<blocks(n8k), EW, 0, stream>>>(kvisH, kcurH, cacheK, Kb, n8k);
    vtprep_kernel<<<blocks(n8k), EW, 0, stream>>>(vvisH, vcurH, cacheV, Vt, n8k);
  }
  attn_flash_kernel<<<NB * NH, 96, 0, stream>>>(qH, Kb, Vt, clen, tmask, attnHb);

  // 5) output projection + residual
  wmma_gemm_kernel<<<g512, 128, 0, stream>>>(attnHb, WoT, bo, attnProj, nullptr, MQ, ND, ND);
  add_inplace_kernel<<<blocks(nOut / 4), EW, 0, stream>>>(combined, attnProj, (int)(nOut / 4));

  // 6) FFN
  ln_kernel<<<MQ, 128, 0, stream>>>(combined, lnfg, lnfb, nullptr, hBf);
  dim3 gffn1(MQ / 32, (4 * ND) / 64);
  wmma_gemm_kernel<<<gffn1, 128, 0, stream>>>(hBf, W1T, b1, ffn1, nullptr, MQ, 4 * ND, ND);
  gelu_cvt_kernel<<<blocks((size_t)MQ * ND), EW, 0, stream>>>(ffn1, ffn1H, MQ * ND);  // MQ*4*ND/4
  wmma_gemm_kernel<<<g512, 128, 0, stream>>>(ffn1H, W2T, b2, ffn2, nullptr, MQ, ND, 4 * ND);

  // 7) finalize: residual, mask, cls select
  finalize_kernel<<<blocks(nOut / 4), EW, 0, stream>>>(combined, ffn2, prev_cls, tmask, act,
                                                       out, ntok, (int)(nOut / 4));

  // 8) cache append K/V
  ln_kernel<<<MT, 128, 0, stream>>>(ntok, lncg, lncb, nullptr, clnH);
  dim3 gapp(MT / 32, ND / 64);
  wmma_gemm_kernel<<<gapp, 128, 0, stream>>>(clnH, WkT, bk, kapp, nullptr, MT, ND, ND);
  wmma_gemm_kernel<<<gapp, 128, 0, stream>>>(clnH, WvT, bv, vapp, nullptr, MT, ND, ND);

  // 9) bulk cache copy then compacting scatter
  hipMemcpyAsync(newK, cacheK, nCache * sizeof(float), hipMemcpyDeviceToDevice, stream);
  hipMemcpyAsync(newV, cacheV, nCache * sizeof(float), hipMemcpyDeviceToDevice, stream);
  dest_kernel<<<NB, 32, 0, stream>>>(tmask, act, clen, dest, lenO);
  scatter_kernel<<<NB * NT, 128, 0, stream>>>(kapp, vapp, dest, newK, newV);
}
